// GNMTDecoder_11089605558603
// MI455X (gfx1250) — matrix operand order repaired and checked by reading, not compiled, not run on hardware
//
#include <hip/hip_runtime.h>
#include <hip/hip_bf16.h>

// ---------------------------------------------------------------------------
// GNMT decoder for gfx1250 (MI455X). All weights converted to bf16 once
// (~131MB total -> L2-resident at 192MB); every GEMM uses
// V_WMMA_F32_16X16X32_BF16 with f32 accumulate. GEMM inner loop is manually
// double-buffered, K-split branch hoisted out of the loop, and each wave owns
// a 16x32 output strip (2 accumulators sharing one A fragment).
// ---------------------------------------------------------------------------

typedef __attribute__((ext_vector_type(16))) __bf16 v16bf;
typedef __attribute__((ext_vector_type(8)))  float  v8f;

union FragBF { v16bf bf; unsigned int u[8]; };

__device__ __forceinline__ unsigned short f2b(float f) {
    unsigned u = __float_as_uint(f);
    unsigned r = 0x7FFFu + ((u >> 16) & 1u);     // round-to-nearest-even
    return (unsigned short)((u + r) >> 16);
}
__device__ __forceinline__ float b2f(unsigned short s) {
    return __uint_as_float(((unsigned)s) << 16);
}
__device__ __forceinline__ float sigm(float x) { return 1.0f / (1.0f + __expf(-x)); }

// A 16x32 bf16 fragment: VGPR v holds K pair {kk, kk+1}, lane group g selects half
__device__ __forceinline__ void load_a(FragBF& f, const unsigned short* p, int g) {
#pragma unroll
    for (int v = 0; v < 8; ++v) {
        int kk = (v < 4) ? (2 * v + 8 * g) : (16 + 2 * (v - 4) + 8 * g);
        f.u[v] = *(const unsigned int*)(p + kk);
    }
}
// B 32x16 bf16 fragment: same K order as load_a (p = row of W[N][K])
__device__ __forceinline__ void load_b(FragBF& f, const unsigned short* p, int g) {
#pragma unroll
    for (int v = 0; v < 8; ++v) f.u[v] = *(const unsigned int*)(p + ((v < 4) ? (2 * v + 8 * g) : (16 + 2 * (v - 4) + 8 * g)));
}

// One straight K segment, software-pipelined: loads for chunk i+1 are issued
// before the WMMAs of chunk i so the waitcnt pass can overlap them.
__device__ __forceinline__ void gemm_segment(const unsigned short* __restrict__ Arow,
                                             const unsigned short* __restrict__ Br0,
                                             const unsigned short* __restrict__ Br1,
                                             int chunks, int g, v8f& acc0, v8f& acc1)
{
    FragBF a, b0, b1;
    load_a(a, Arow, g);
    load_b(b0, Br0, g);
    load_b(b1, Br1, g);
    for (int i = 1; i < chunks; ++i) {
        FragBF an, c0, c1;
        load_a(an, Arow + (size_t)i * 32, g);
        load_b(c0, Br0 + (size_t)i * 32, g);
        load_b(c1, Br1 + (size_t)i * 32, g);
        acc0 = __builtin_amdgcn_wmma_f32_16x16x32_bf16(false, a.bf, false, b0.bf,
                                                       (short)0, acc0, false, false);
        acc1 = __builtin_amdgcn_wmma_f32_16x16x32_bf16(false, a.bf, false, b1.bf,
                                                       (short)0, acc1, false, false);
        a = an; b0 = c0; b1 = c1;
    }
    acc0 = __builtin_amdgcn_wmma_f32_16x16x32_bf16(false, a.bf, false, b0.bf,
                                                   (short)0, acc0, false, false);
    acc1 = __builtin_amdgcn_wmma_f32_16x16x32_bf16(false, a.bf, false, b1.bf,
                                                   (short)0, acc1, false, false);
}

// ---------------------------------------------------------------------------
// C[M x N] = A[M x K] * B^T (+ bias[n]).  A bf16 row-major, optionally split
// at kSplit between A0 / A1 (x || h).  B = W[N][K] bf16 row-major.
// Block = 256 threads (8 waves); wave -> (m-tile = wave&1, 32-wide n-strip =
// wave>>1), so a block covers 32 x 128 of C.
// ---------------------------------------------------------------------------
__global__ void wmma_gemm(const unsigned short* __restrict__ A0,
                          const unsigned short* __restrict__ A1,
                          int kSplit, int K, int lda,
                          const unsigned short* __restrict__ B,
                          const float* __restrict__ bias,
                          float* __restrict__ Cf,
                          unsigned short* __restrict__ Cb,
                          long ldc, long aStride, long cStride)
{
    const int lane = threadIdx.x & 31;
    const int wave = threadIdx.x >> 5;
    const int g    = lane >> 4;
    const int hl   = lane & 15;
    const int nB   = blockIdx.x * 128 + (wave >> 1) * 32;   // wave's n-strip
    const int n0   = nB + hl;
    const int n1   = nB + 16 + hl;
    const int m0   = blockIdx.y * 32 + (wave & 1) * 16;

    const unsigned short* Brow0 = B + (size_t)n0 * K;
    const unsigned short* Brow1 = B + (size_t)n1 * K;

    v8f acc0 = {}, acc1 = {};

    const unsigned short* Arow0 =
        A0 + (size_t)blockIdx.z * aStride + (size_t)(m0 + hl) * lda;
    gemm_segment(Arow0, Brow0, Brow1, kSplit / 32, g, acc0, acc1);

    if (K > kSplit) {                                       // second A operand (h)
        const unsigned short* Arow1 = A1 + (size_t)(m0 + hl) * lda;
        gemm_segment(Arow1, Brow0 + kSplit, Brow1 + kSplit,
                     (K - kSplit) / 32, g, acc0, acc1);
    }

    const float bv0 = bias ? bias[n0] : 0.0f;
    const float bv1 = bias ? bias[n1] : 0.0f;
    if (Cf) {
        float* Cp = Cf + (size_t)blockIdx.z * cStride;
#pragma unroll
        for (int r = 0; r < 8; ++r) {
            int m = m0 + r + 8 * g;                         // C/D layout: M = r + 8g
            Cp[(size_t)m * ldc + n0] = acc0[r] + bv0;
            Cp[(size_t)m * ldc + n1] = acc1[r] + bv1;
        }
    } else {
        unsigned short* Cp = Cb + (size_t)blockIdx.z * cStride;
#pragma unroll
        for (int r = 0; r < 8; ++r) {
            int m = m0 + r + 8 * g;
            Cp[(size_t)m * ldc + n0] = f2b(acc0[r] + bv0);
            Cp[(size_t)m * ldc + n1] = f2b(acc1[r] + bv1);
        }
    }
}

// ------------------------- pre-pass conversion kernels ---------------------

__global__ void convert_f2b(const float* __restrict__ src,
                            unsigned short* __restrict__ dst, long n)
{
    long i = (long)blockIdx.x * blockDim.x + threadIdx.x;
    if (i < n) dst[i] = f2b(src[i]);
}

// Fuse [Wih | Whh] -> [L][4096][2048] bf16 so each LSTM layer is one GEMM.
__global__ void build_wfused(const float* __restrict__ wih,
                             const float* __restrict__ whh,
                             unsigned short* __restrict__ dst)
{
    long i  = (long)blockIdx.x * blockDim.x + threadIdx.x;  // over 4*4096*2048
    int  k  = (int)(i & 2047);
    long nl = i >> 11;                                      // l*4096 + n
    float v = (k < 1024) ? wih[nl * 1024 + k] : whh[nl * 1024 + (k - 1024)];
    dst[i] = f2b(v);
}

__global__ void bias_and_init(const float* __restrict__ bih,
                              const float* __restrict__ bhh,
                              float* __restrict__ biasSum,
                              float* __restrict__ c_f,
                              unsigned short* __restrict__ h_b,
                              float* __restrict__ ctx_f)
{
    int i = blockIdx.x * blockDim.x + threadIdx.x;          // 131072 threads
    if (i < 16384)  biasSum[i] = bih[i] + bhh[i];
    if (i < 131072) { c_f[i] = 0.0f; h_b[i] = 0; }
    if (i < 16384)  ctx_f[i] = 0.0f;
}

// ------------------------- per-step kernels --------------------------------

__global__ void embed_concat(const int* __restrict__ tgt,
                             const float* __restrict__ emb,
                             const float* __restrict__ ctx,
                             float* __restrict__ stepin_f,
                             unsigned short* __restrict__ stepin_b, int t)
{
    int i = blockIdx.x * blockDim.x + threadIdx.x;          // 32*1024
    int b = i >> 10, j = i & 1023;
    float v = (j < 512) ? emb[(size_t)tgt[b * 64 + t] * 512 + j]
                        : ctx[b * 512 + (j - 512)];
    stepin_f[i] = v;
    stepin_b[i] = f2b(v);
}

__global__ void lstm_cell(const float* __restrict__ G,
                          float* __restrict__ c,
                          unsigned short* __restrict__ hB,
                          unsigned short* __restrict__ xB,
                          const float* __restrict__ stepin, int residual)
{
    int i = blockIdx.x * blockDim.x + threadIdx.x;          // 32*1024
    int b = i >> 10, j = i & 1023;
    const float* gr = G + (size_t)b * 4096;                 // torch order i,f,g,o
    float ig = sigm(gr[j]);
    float fg = sigm(gr[1024 + j]);
    float gg = tanhf(gr[2048 + j]);
    float og = sigm(gr[3072 + j]);
    float cc = fg * c[i] + ig * gg;
    float hh = og * tanhf(cc);
    c[i]  = cc;
    hB[i] = f2b(hh);
    float x = residual ? (hh + stepin[i]) : hh;
    xB[i] = f2b(x);
}

// Luong attention for q-len 1: scores (scaled, masked) -> softmax -> context.
// One block per batch row, 128 threads (= TS).
__global__ void attn_kernel(const float* __restrict__ qf,
                            const unsigned short* __restrict__ kproj,
                            const int* __restrict__ mask,
                            const float* __restrict__ enc,
                            float* __restrict__ ctx)
{
    const int b = blockIdx.x, s = threadIdx.x;
    __shared__ float sq[1024];
    __shared__ float red[128];
    __shared__ float wb[128];
    for (int d = s; d < 1024; d += 128) sq[d] = qf[b * 1024 + d];
    __syncthreads();

    const unsigned short* kp = kproj + ((size_t)b * 128 + s) * 1024;
    float acc = 0.0f;
#pragma unroll 4
    for (int d = 0; d < 1024; d += 2) {
        unsigned pk = *(const unsigned int*)(kp + d);
        acc = fmaf(sq[d],     __uint_as_float(pk << 16),          acc);
        acc = fmaf(sq[d + 1], __uint_as_float(pk & 0xFFFF0000u),  acc);
    }
    float score = acc * 0.03125f;                     // 1/sqrt(1024)
    if (mask[b * 128 + s] == 0) score = -10000.0f;

    red[s] = score; __syncthreads();
    for (int o = 64; o; o >>= 1) { if (s < o) red[s] = fmaxf(red[s], red[s + o]); __syncthreads(); }
    float mx = red[0]; __syncthreads();
    float e = __expf(score - mx);
    red[s] = e; __syncthreads();
    for (int o = 64; o; o >>= 1) { if (s < o) red[s] += red[s + o]; __syncthreads(); }
    float w = e / red[0];
    wb[s] = w; __syncthreads();

    for (int e0 = s; e0 < 512; e0 += 128) {
        float a = 0.0f;
        const float* ep = enc + ((size_t)b * 128) * 512 + e0;
#pragma unroll 4
        for (int ss = 0; ss < 128; ++ss) a = fmaf(wb[ss], ep[(size_t)ss * 512], a);
        ctx[b * 512 + e0] = a;
    }
}

// ---------------------------------------------------------------------------

extern "C" void kernel_launch(void* const* d_in, const int* in_sizes, int n_in,
                              void* d_out, int out_size, void* d_ws, size_t ws_size,
                              hipStream_t stream)
{
    (void)in_sizes; (void)n_in; (void)out_size; (void)ws_size;
    const int V = 32000, L = 4, B = 32, TT = 64;

    const int*   tgt = (const int*)  d_in[0];
    const float* enc = (const float*)d_in[2];
    const int*   msk = (const int*)  d_in[3];
    const float* emb = (const float*)d_in[4];
    const float* wih = (const float*)d_in[5];
    const float* whh = (const float*)d_in[6];
    const float* bih = (const float*)d_in[7];
    const float* bhh = (const float*)d_in[8];
    const float* Wq  = (const float*)d_in[9];
    const float* Wk  = (const float*)d_in[10];
    const float* Pw  = (const float*)d_in[11];
    const float* Pb  = (const float*)d_in[12];
    float* out = (float*)d_out;

    // ---- workspace plan (~150 MB, 256B aligned slices) ----
    char*  ws  = (char*)d_ws;
    size_t off = 0;
    auto alloc = [&](size_t bytes) -> char* {
        char* p = ws + off;
        off = (off + bytes + 255) & ~(size_t)255;
        return p;
    };
    unsigned short* WfusedB = (unsigned short*)alloc((size_t)L * 4096 * 2048 * 2); // 64 MB
    unsigned short* WqB     = (unsigned short*)alloc((size_t)1024 * 1024 * 2);
    unsigned short* PwB     = (unsigned short*)alloc((size_t)V * 1024 * 2);        // 62.5 MB
    unsigned short* WkB     = (unsigned short*)alloc((size_t)1024 * 512 * 2);
    unsigned short* EncB    = (unsigned short*)alloc((size_t)B * 128 * 512 * 2);
    unsigned short* KprojB  = (unsigned short*)alloc((size_t)B * 128 * 1024 * 2);  // 8 MB
    unsigned short* hB      = (unsigned short*)alloc((size_t)L * B * 1024 * 2);
    unsigned short* stepinB = (unsigned short*)alloc((size_t)B * 1024 * 2);
    unsigned short* xB      = (unsigned short*)alloc((size_t)B * 1024 * 2);
    float* cF      = (float*)alloc((size_t)L * B * 1024 * 4);
    float* stepinF = (float*)alloc((size_t)B * 1024 * 4);
    float* Gf      = (float*)alloc((size_t)B * 4096 * 4);
    float* qF      = (float*)alloc((size_t)B * 1024 * 4);
    float* ctxF    = (float*)alloc((size_t)B * 512 * 4);
    float* biasSum = (float*)alloc((size_t)L * 4096 * 4);

    // ---- pre-pass: bf16 weight conversion + state init + k_proj ----
    build_wfused<<<(L * 4096 * 2048) / 256, 256, 0, stream>>>(wih, whh, WfusedB);
    convert_f2b<<<(1024 * 1024) / 256, 256, 0, stream>>>(Wq, WqB, 1024 * 1024);
    convert_f2b<<<((long)V * 1024) / 256, 256, 0, stream>>>(Pw, PwB, (long)V * 1024);
    convert_f2b<<<(1024 * 512) / 256, 256, 0, stream>>>(Wk, WkB, 1024 * 512);
    convert_f2b<<<(B * 128 * 512) / 256, 256, 0, stream>>>(enc, EncB, (long)B * 128 * 512);
    bias_and_init<<<512, 256, 0, stream>>>(bih, bhh, biasSum, cF, hB, ctxF);

    // k_proj[b] = enc[b] (128x512) @ Wk^T (512x1024), bf16 output, per-batch
    wmma_gemm<<<dim3(1024 / 128, 128 / 32, B), 256, 0, stream>>>(
        EncB, nullptr, /*kSplit=*/512, /*K=*/512, /*lda=*/512,
        WkB, nullptr, nullptr, KprojB,
        /*ldc=*/1024, /*aStride=*/128 * 512, /*cStride=*/128 * 1024);

    // ---- 64 sequential decode steps ----
    for (int t = 0; t < TT; ++t) {
        embed_concat<<<(B * 1024) / 256, 256, 0, stream>>>(tgt, emb, ctxF,
                                                           stepinF, stepinB, t);
        for (int l = 0; l < L; ++l) {
            const unsigned short* a0 = (l == 0) ? stepinB : xB;
            // G = [x | h_l] @ [Wih_l | Whh_l]^T + (b_ih + b_hh)
            wmma_gemm<<<dim3(4096 / 128, 1, 1), 256, 0, stream>>>(
                a0, hB + (size_t)l * B * 1024, /*kSplit=*/1024, /*K=*/2048, /*lda=*/1024,
                WfusedB + (size_t)l * 4096 * 2048, biasSum + (size_t)l * 4096,
                Gf, nullptr, /*ldc=*/4096, 0, 0);
            lstm_cell<<<(B * 1024) / 256, 256, 0, stream>>>(
                Gf, cF + (size_t)l * B * 1024, hB + (size_t)l * B * 1024,
                xB, stepinF, (l >= 2) ? 1 : 0);
        }
        // q = x @ Wq^T  (f32 out)
        wmma_gemm<<<dim3(1024 / 128, 1, 1), 256, 0, stream>>>(
            xB, nullptr, 1024, 1024, 1024, WqB, nullptr, qF, nullptr, 1024, 0, 0);
        // attention -> ctx for next step
        attn_kernel<<<B, 128, 0, stream>>>(qF, KprojB, msk, enc, ctxF);
        // logits[:, t, :] = x @ proj_w^T + proj_b   (independent of attention)
        wmma_gemm<<<dim3(V / 128, 1, 1), 256, 0, stream>>>(
            xB, nullptr, 1024, 1024, 1024, PwB, Pb,
            out + (size_t)t * V, nullptr, /*ldc=*/(long)TT * V, 0, 0);
    }
}
